// ScaledDotProductAdaptiveAttention_14181982011822
// MI455X (gfx1250) — matrix operand order repaired and checked by reading, not compiled, not run on hardware
//
#include <hip/hip_runtime.h>
#include <hip/hip_bf16.h>
#include <math.h>

#define D_MODEL 1024
#define HEADS   16
#define DHEAD   64
#define BATCH   4
#define NQQ     2048
#define NKK     2048
#define MROWS   (BATCH * NQQ)      // 8192
#define QSCALE  0.125f             // 1/sqrt(64)

typedef __attribute__((ext_vector_type(16))) __bf16 v16bf;
typedef __attribute__((ext_vector_type(8)))  float  v8f;
typedef __attribute__((ext_vector_type(2)))  float  v2f;
typedef __attribute__((ext_vector_type(4)))  int    v4i;

// CDNA5 async global->LDS copy engine (ASYNCcnt) if the toolchain exposes it.
#if defined(__has_builtin)
# if __has_builtin(__builtin_amdgcn_global_load_async_to_lds_b128) && \
     __has_builtin(__builtin_amdgcn_s_wait_asynccnt)
#  define USE_ASYNC 1
# endif
#endif
#ifndef USE_ASYNC
# define USE_ASYNC 0
#endif

#if USE_ASYNC
#define ASYNC_WAIT() __builtin_amdgcn_s_wait_asynccnt(0)
__device__ __forceinline__ void async_cp16(const unsigned short* g, unsigned short* l) {
  __builtin_amdgcn_global_load_async_to_lds_b128(
      (__attribute__((address_space(1))) v4i*)g,
      (__attribute__((address_space(3))) v4i*)l, 0, 0);
}
#else
#define ASYNC_WAIT()
__device__ __forceinline__ void async_cp16(const unsigned short* g, unsigned short* l) {
  *(uint4*)l = *(const uint4*)g;
}
#endif

__device__ __forceinline__ unsigned short f2bf(float f) {
  union { float f; unsigned u; } v; v.f = f;
  unsigned r = v.u + 0x7FFFu + ((v.u >> 16) & 1u);   // RNE
  return (unsigned short)(r >> 16);
}
__device__ __forceinline__ float bf2f(unsigned short b) {
  union { unsigned u; float f; } v; v.u = ((unsigned)b) << 16; return v.f;
}

union Frag16 { uint4 u4[2]; v16bf v; };   // 16 bf16 = 8 VGPRs

// ---------------------------------------------------------------------------
// Weight transpose + fp32 -> bf16:  W (K x N, row major) -> WT (N x K) bf16
// ---------------------------------------------------------------------------
__global__ __launch_bounds__(256)
void transpose_w_bf16(const float* __restrict__ W, unsigned short* __restrict__ WT,
                      int K, int N) {
  __shared__ float t[32][33];
  const int bx = blockIdx.x * 32;   // n
  const int by = blockIdx.y * 32;   // k
  const int tx = threadIdx.x & 31, ty = threadIdx.x >> 5;  // 32 x 8
  #pragma unroll
  for (int i = 0; i < 32; i += 8)
    t[ty + i][tx] = W[(size_t)(by + ty + i) * N + bx + tx];
  __syncthreads();
  #pragma unroll
  for (int i = 0; i < 32; i += 8)
    WT[(size_t)(bx + ty + i) * K + by + tx] = f2bf(t[tx][ty + i]);
}

// ---------------------------------------------------------------------------
// Tiled WMMA GEMM: C[M,N] = out_scale * (A[M,K] @ WT[N,K]^T + bias[N])
// BM=BN=128, BK=32, 256 threads = 8 waves, each wave 64x32 via 4x2 WMMA tiles.
// Double-buffered LDS; B tile (and bf16 A tile) staged via async copies that
// overlap the WMMA block; fp32 A tile uses load-early/convert-late software
// pipelining around the WMMAs.
// ---------------------------------------------------------------------------
template<bool A_F32, bool OUT_F32>
__global__ __launch_bounds__(256)
void gemm_wmma_kernel(const void* __restrict__ Ap, const unsigned short* __restrict__ WT,
                      const float* __restrict__ bias, void* __restrict__ Cp,
                      int M, int N, int K, float out_scale) {
  __shared__ unsigned short As[2][128 * 32];
  __shared__ unsigned short Bs[2][128 * 32];
  const int tid  = threadIdx.x;
  const int lane = tid & 31, wid = tid >> 5;
  const int wrow = wid >> 2, wcol = wid & 3;        // 2 x 4 wave grid
  const int m0 = blockIdx.y * 128, n0 = blockIdx.x * 128;
  const int lh = lane >> 4, lm = lane & 15;
  const int row = tid >> 1, half = tid & 1;         // staging assignment

  v8f acc[4][2];
  #pragma unroll
  for (int i = 0; i < 4; ++i)
    #pragma unroll
    for (int j = 0; j < 2; ++j) acc[i][j] = (v8f){0,0,0,0,0,0,0,0};

  float4 ar[4];   // fp32-A pipeline registers

  auto loadA = [&](int kt) {   // fp32 path: global -> regs
    if (A_F32) {
      const float4* src = (const float4*)((const float*)Ap +
                          (size_t)(m0 + row) * K + (kt << 5) + half * 16);
      #pragma unroll
      for (int j = 0; j < 4; ++j) ar[j] = src[j];
    }
  };
  auto storeA = [&](int buf) { // fp32 path: convert -> LDS
    if (A_F32) {
      unsigned* dst = (unsigned*)&As[buf][row * 32 + half * 16];
      #pragma unroll
      for (int j = 0; j < 4; ++j) {
        dst[2*j+0] = (unsigned)f2bf(ar[j].x) | ((unsigned)f2bf(ar[j].y) << 16);
        dst[2*j+1] = (unsigned)f2bf(ar[j].z) | ((unsigned)f2bf(ar[j].w) << 16);
      }
    }
  };
  auto stageAsync = [&](int kt, int buf) {  // async engine: B tile (+ bf16 A tile)
    const int k0 = kt << 5;
    if (!A_F32) {
      const unsigned short* g = (const unsigned short*)Ap +
                                (size_t)(m0 + row) * K + k0 + half * 16;
      async_cp16(g,     &As[buf][row * 32 + half * 16]);
      async_cp16(g + 8, &As[buf][row * 32 + half * 16 + 8]);
    }
    const unsigned short* gb = WT + (size_t)(n0 + row) * K + k0 + half * 16;
    async_cp16(gb,     &Bs[buf][row * 32 + half * 16]);
    async_cp16(gb + 8, &Bs[buf][row * 32 + half * 16 + 8]);
  };

  const int ktiles = K >> 5;
  loadA(0); storeA(0);
  stageAsync(0, 0);
  ASYNC_WAIT();
  __syncthreads();

  for (int kt = 0; kt < ktiles; ++kt) {
    const int buf = kt & 1;
    const bool more = (kt + 1 < ktiles);
    if (more) {                      // prefetch next tile into other buffer
      loadA(kt + 1);
      stageAsync(kt + 1, buf ^ 1);
      if (A_F32 && kt + 2 < ktiles)  // speculative L2 warm-up -> global_prefetch_b8
        __builtin_prefetch((const float*)Ap + (size_t)(m0 + row) * K +
                           ((kt + 2) << 5) + half * 16, 0, 1);
    }

    const uint4* As4 = (const uint4*)As[buf];  // row = 4 uint4 units (8 bf16 each)
    const uint4* Bs4 = (const uint4*)Bs[buf];
    Frag16 b[2];
    #pragma unroll
    for (int jj = 0; jj < 2; ++jj) {           // B frag: lanes<16 K0-15, >=16 K16-31
      const int n = wcol * 32 + jj * 16 + lm;
      b[jj].u4[0] = Bs4[n * 4 + lh * 2 + 0];
      b[jj].u4[1] = Bs4[n * 4 + lh * 2 + 1];
    }
    #pragma unroll
    for (int i = 0; i < 4; ++i) {              // A frag: lanes<16 K0-15, >=16 K16-31
      const int r = wrow * 64 + i * 16 + lm;
      Frag16 a;
      a.u4[0] = As4[r * 4 + lh * 2 + 0];
      a.u4[1] = As4[r * 4 + lh * 2 + 1];
      #pragma unroll
      for (int jj = 0; jj < 2; ++jj)
        acc[i][jj] = __builtin_amdgcn_wmma_f32_16x16x32_bf16(
            false, a.v, false, b[jj].v, (short)0, acc[i][jj], false, false);
    }
    if (more) storeA(buf ^ 1);      // convert lands after the WMMA block
    ASYNC_WAIT();
    __syncthreads();
  }

  // epilogue: C tile layout -> VGPR v holds rows v (lanes 0-15) and v+8 (16-31)
  #pragma unroll
  for (int i = 0; i < 4; ++i)
    #pragma unroll
    for (int jj = 0; jj < 2; ++jj) {
      const int n = n0 + wcol * 32 + jj * 16 + lm;
      const float bv = bias[n];
      #pragma unroll
      for (int v = 0; v < 8; ++v) {
        const int m = m0 + wrow * 64 + i * 16 + v + lh * 8;
        const float val = (acc[i][jj][v] + bv) * out_scale;
        if (OUT_F32) ((float*)Cp)[(size_t)m * N + n] = val;
        else ((unsigned short*)Cp)[(size_t)m * N + n] = f2bf(val);
      }
    }
}

// ---------------------------------------------------------------------------
// KtV[b,h] = K_h^T @ V_h  (64x64 fp32 per head), reduction over NK=2048.
// One workgroup (8 waves) per (b,h); K/V chunks staged transposed in LDS.
// ---------------------------------------------------------------------------
__global__ __launch_bounds__(256)
void ktv_kernel(const unsigned short* __restrict__ Kw, const unsigned short* __restrict__ Vw,
                float* __restrict__ KtV) {
  const int bh = blockIdx.x, b = bh >> 4, h = bh & 15;
  const size_t rowbase = (size_t)b * NKK;
  const int tid = threadIdx.x, lane = tid & 31, wid = tid >> 5;
  const int lh = lane >> 4, lm = lane & 15;
  const int t0 = wid * 2;
  const int r  = t0 >> 2;                  // d1 tile (shared by this wave's pair)
  const int c0 = t0 & 3, c1 = c0 + 1;      // d2 tiles
  __shared__ unsigned short KT[64 * 40];   // [d][n], stride 40 (80B, 16B aligned)
  __shared__ unsigned short VT[64 * 40];
  v8f acc0 = (v8f){0,0,0,0,0,0,0,0}, acc1 = acc0;

  for (int n0 = 0; n0 < NKK; n0 += 32) {
    const int d = tid & 63, nb = tid >> 6;
    #pragma unroll
    for (int s = 0; s < 8; ++s) {
      const int nn = nb + s * 4;
      const size_t g = (rowbase + n0 + nn) * (size_t)D_MODEL + h * DHEAD + d;
      KT[d * 40 + nn] = Kw[g];
      VT[d * 40 + nn] = Vw[g];
    }
    __syncthreads();
    const uint4* KT4 = (const uint4*)KT;   // 5 uint4 units per row
    const uint4* VT4 = (const uint4*)VT;
    const int d1 = r * 16 + lm;
    Frag16 a, bb0, bb1;
    a.u4[0] = KT4[d1 * 5 + lh * 2 + 0];
    a.u4[1] = KT4[d1 * 5 + lh * 2 + 1];
    const int e0 = c0 * 16 + lm, e1 = c1 * 16 + lm;
    bb0.u4[0] = VT4[e0 * 5 + lh * 2 + 0];  bb0.u4[1] = VT4[e0 * 5 + lh * 2 + 1];
    bb1.u4[0] = VT4[e1 * 5 + lh * 2 + 0];  bb1.u4[1] = VT4[e1 * 5 + lh * 2 + 1];
    acc0 = __builtin_amdgcn_wmma_f32_16x16x32_bf16(false, a.v, false, bb0.v, (short)0, acc0, false, false);
    acc1 = __builtin_amdgcn_wmma_f32_16x16x32_bf16(false, a.v, false, bb1.v, (short)0, acc1, false, false);
    __syncthreads();
  }
  float* out = KtV + (size_t)bh * (DHEAD * DHEAD);
  #pragma unroll
  for (int v = 0; v < 8; ++v) {
    const int d1 = r * 16 + v + lh * 8;
    out[d1 * 64 + c0 * 16 + lm] = acc0[v];
    out[d1 * 64 + c1 * 16 + lm] = acc1[v];
  }
}

// ---------------------------------------------------------------------------
// Attention: per (b,h,64-q-tile) workgroup (4 waves, 16 q rows each).
// Streams all keys with WMMA logits + per-lane online logsumexp (no V accum),
// with the next 32-key chunk async-prefetched into the alternate LDS buffer.
// Then beta from the lang term, att_v = Qs @ KtV via v_wmma_f32_16x16x4_f32,
// blend with S, write bf16 heads.
// ---------------------------------------------------------------------------
__global__ __launch_bounds__(128)
void attn_kernel(const unsigned short* __restrict__ Qw, const unsigned short* __restrict__ Kw,
                 const unsigned short* __restrict__ Sw, const float* __restrict__ KtV,
                 unsigned short* __restrict__ Ow) {
  const int qt = blockIdx.x, h = blockIdx.y, b = blockIdx.z;
  const int q0 = qt * 64;
  const int tid = threadIdx.x, lane = tid & 31, wid = tid >> 5;
  const int lh = lane >> 4, lm = lane & 15;
  __shared__ unsigned short Kt[2][32 * 64];  // double-buffered key chunk [key][kd]
  __shared__ float Qs[64 * 64];              // scaled Q tile, fp32
  __shared__ float Ss[64 * 64];              // S tile, fp32
  __shared__ float Cs[64 * 64];              // KtV[b,h], fp32
  const size_t qrow0 = (size_t)b * NQQ + q0;
  const size_t krow0 = (size_t)b * NKK;

  auto stageK = [&](int kb, int buf) {       // 256 x 16B units, async engine
    #pragma unroll
    for (int u = tid; u < 256; u += 128) {
      const int rr = u >> 3, d8 = (u & 7) * 8;
      async_cp16(&Kw[(krow0 + kb + rr) * (size_t)D_MODEL + h * DHEAD + d8],
                 &Kt[buf][rr * 64 + d8]);
    }
  };

  stageK(0, 0);
  for (int idx = tid; idx < 64 * 64; idx += 128) {
    const int rr = idx >> 6, dd = idx & 63;
    const size_t g = (qrow0 + rr) * (size_t)D_MODEL + h * DHEAD + dd;
    Qs[idx] = bf2f(Qw[g]);
    Ss[idx] = bf2f(Sw[g]);
    Cs[idx] = KtV[((size_t)(b * HEADS + h)) * (DHEAD * DHEAD) + idx];
  }
  ASYNC_WAIT();
  __syncthreads();

  // per-wave Q fragments (16x64 -> 2x v16bf, contiguous k = 16*lh + e per 32-k slice)
  Frag16 fa[2];
  const int qr  = wid * 16 + lm;
  const int klo = lh * 8;
  #pragma unroll
  for (int f = 0; f < 2; ++f)
    #pragma unroll
    for (int e = 0; e < 16; ++e) {
      const int kd = f * 32 + lh * 16 + e;
      fa[f].v[e] = (__bf16)Qs[qr * 64 + kd];
    }

  float m[8], l[8];
  #pragma unroll
  for (int v = 0; v < 8; ++v) { m[v] = -INFINITY; l[v] = 0.f; }

  for (int kb = 0; kb < NKK; kb += 32) {
    const int buf = (kb >> 5) & 1;
    if (kb + 32 < NKK) stageK(kb + 32, buf ^ 1);   // overlap next chunk
    const uint4* Kt4 = (const uint4*)Kt[buf];      // 8 uint4 units per key row
    #pragma unroll
    for (int sub = 0; sub < 2; ++sub) {
      const int keyr = sub * 16 + lm;
      Frag16 fb0, fb1;
      fb0.u4[0] = Kt4[keyr * 8 + 0 + lh * 2];  fb0.u4[1] = Kt4[keyr * 8 + 1 + lh * 2];
      fb1.u4[0] = Kt4[keyr * 8 + 4 + lh * 2];  fb1.u4[1] = Kt4[keyr * 8 + 5 + lh * 2];
      v8f cc = (v8f){0,0,0,0,0,0,0,0};
      cc = __builtin_amdgcn_wmma_f32_16x16x32_bf16(false, fa[0].v, false, fb0.v, (short)0, cc, false, false);
      cc = __builtin_amdgcn_wmma_f32_16x16x32_bf16(false, fa[1].v, false, fb1.v, (short)0, cc, false, false);
      #pragma unroll
      for (int v = 0; v < 8; ++v) {          // per-lane online logsumexp
        const float x  = cc[v];
        const float nm = fmaxf(m[v], x);
        l[v] = l[v] * __expf(m[v] - nm) + __expf(x - nm);
        m[v] = nm;
      }
    }
    ASYNC_WAIT();
    __syncthreads();
  }

  // merge (m,l) across the 16 lanes of each half-wave (keys are column-striped)
  #pragma unroll
  for (int v = 0; v < 8; ++v)
    #pragma unroll
    for (int off = 1; off < 16; off <<= 1) {
      const float om = __shfl_xor(m[v], off, 32);
      const float ol = __shfl_xor(l[v], off, 32);
      const float nm = fmaxf(m[v], om);
      l[v] = l[v] * __expf(m[v] - nm) + ol * __expf(om - nm);
      m[v] = nm;
    }

  // lang term: scale*q . s per row (Q already scaled); split dims across halves
  float part = 0.f;
  const int lrow = wid * 16 + lm;
  #pragma unroll
  for (int dd = 0; dd < 32; ++dd)
    part += Qs[lrow * 64 + lh * 32 + dd] * Ss[lrow * 64 + lh * 32 + dd];
  part += __shfl_xor(part, 16, 32);        // lane p holds lang of row p%16

  float beta[8];
  #pragma unroll
  for (int v = 0; v < 8; ++v) {
    const int rr = v + lh * 8;
    const float lg = __shfl(part, (lane & 16) | rr, 32);
    const float nm = fmaxf(m[v], lg);
    const float lf = l[v] * __expf(m[v] - nm) + __expf(lg - nm);
    beta[v] = __expf(lg - nm) / lf;
  }

  // att_v = Qs(16x64) @ KtV(64x64) in full fp32 WMMA, then blend + store
  #pragma unroll
  for (int d2t = 0; d2t < 4; ++d2t) {
    v8f av = (v8f){0,0,0,0,0,0,0,0};
#if __has_builtin(__builtin_amdgcn_wmma_f32_16x16x4_f32)
    #pragma unroll
    for (int k0 = 0; k0 < 64; k0 += 4) {
      const int ks = k0 + lh * 2;          // f32 frag: lanes<16 K0,1; >=16 K2,3
      v2f a2, b2;
      a2[0] = Qs[(wid * 16 + lm) * 64 + ks];
      a2[1] = Qs[(wid * 16 + lm) * 64 + ks + 1];
      b2[0] = Cs[ks * 64 + d2t * 16 + lm];
      b2[1] = Cs[(ks + 1) * 64 + d2t * 16 + lm];
      av = __builtin_amdgcn_wmma_f32_16x16x4_f32(false, a2, false, b2, (short)0, av, false, false);
    }
#else
    #pragma unroll
    for (int v = 0; v < 8; ++v) {
      const int rr = v + lh * 8;
      float s = 0.f;
      for (int kk = 0; kk < 64; ++kk)
        s += Qs[(wid * 16 + rr) * 64 + kk] * Cs[kk * 64 + d2t * 16 + lm];
      av[v] = s;
    }
#endif
    #pragma unroll
    for (int v = 0; v < 8; ++v) {
      const int rr = v + lh * 8;
      const int dd = d2t * 16 + lm;
      const float o = beta[v] * Ss[(wid * 16 + rr) * 64 + dd] + (1.f - beta[v]) * av[v];
      Ow[(qrow0 + wid * 16 + rr) * (size_t)D_MODEL + h * DHEAD + dd] = f2bf(o);
    }
  }
}

// ---------------------------------------------------------------------------
extern "C" void kernel_launch(void* const* d_in, const int* in_sizes, int n_in,
                              void* d_out, int out_size, void* d_ws, size_t ws_size,
                              hipStream_t stream) {
  const float* queries = (const float*)d_in[0];
  const float* keys    = (const float*)d_in[1];
  const float* values  = (const float*)d_in[2];
  const float* langf   = (const float*)d_in[3];
  const float* Wq = (const float*)d_in[4];   const float* bq = (const float*)d_in[5];
  const float* Wk = (const float*)d_in[6];   const float* bk = (const float*)d_in[7];
  const float* Wv = (const float*)d_in[8];   const float* bv = (const float*)d_in[9];
  const float* Wsm = (const float*)d_in[10]; const float* bs = (const float*)d_in[11];
  const float* Wo = (const float*)d_in[12];  const float* bo = (const float*)d_in[13];
  (void)in_sizes; (void)n_in; (void)out_size; (void)ws_size;

  char* ws = (char*)d_ws;
  size_t off = 0;
  auto alloc = [&](size_t bytes) {
    char* p = ws + off;
    off = (off + bytes + 255) & ~(size_t)255;
    return p;
  };
  const size_t WB = (size_t)D_MODEL * D_MODEL * sizeof(unsigned short);
  const size_t MB = (size_t)MROWS * D_MODEL * sizeof(unsigned short);
  unsigned short* WqT = (unsigned short*)alloc(WB);
  unsigned short* WkT = (unsigned short*)alloc(WB);
  unsigned short* WvT = (unsigned short*)alloc(WB);
  unsigned short* WsT = (unsigned short*)alloc(WB);
  unsigned short* WoT = (unsigned short*)alloc(WB);
  unsigned short* Qws = (unsigned short*)alloc(MB);
  unsigned short* Kws = (unsigned short*)alloc(MB);
  unsigned short* Vws = (unsigned short*)alloc(MB);
  unsigned short* Sws = (unsigned short*)alloc(MB);
  unsigned short* Ows = (unsigned short*)alloc(MB);
  float* KtVws = (float*)alloc((size_t)BATCH * HEADS * DHEAD * DHEAD * sizeof(float));

  dim3 tg(D_MODEL / 32, D_MODEL / 32);
  transpose_w_bf16<<<tg, 256, 0, stream>>>(Wq,  WqT, D_MODEL, D_MODEL);
  transpose_w_bf16<<<tg, 256, 0, stream>>>(Wk,  WkT, D_MODEL, D_MODEL);
  transpose_w_bf16<<<tg, 256, 0, stream>>>(Wv,  WvT, D_MODEL, D_MODEL);
  transpose_w_bf16<<<tg, 256, 0, stream>>>(Wsm, WsT, D_MODEL, D_MODEL);
  transpose_w_bf16<<<tg, 256, 0, stream>>>(Wo,  WoT, D_MODEL, D_MODEL);

  dim3 gp(D_MODEL / 128, MROWS / 128);   // (N tiles, M tiles)
  gemm_wmma_kernel<true, false><<<gp, 256, 0, stream>>>(queries, WqT, bq, Qws, MROWS, D_MODEL, D_MODEL, QSCALE);
  gemm_wmma_kernel<true, false><<<gp, 256, 0, stream>>>(keys,    WkT, bk, Kws, MROWS, D_MODEL, D_MODEL, 1.f);
  gemm_wmma_kernel<true, false><<<gp, 256, 0, stream>>>(values,  WvT, bv, Vws, MROWS, D_MODEL, D_MODEL, 1.f);
  gemm_wmma_kernel<true, false><<<gp, 256, 0, stream>>>(langf,   WsT, bs, Sws, MROWS, D_MODEL, D_MODEL, 1.f);

  ktv_kernel<<<BATCH * HEADS, 256, 0, stream>>>(Kws, Vws, KtVws);

  dim3 ga(NQQ / 64, HEADS, BATCH);
  attn_kernel<<<ga, 128, 0, stream>>>(Qws, Kws, Sws, KtVws, Ows);

  gemm_wmma_kernel<false, true><<<gp, 256, 0, stream>>>(Ows, WoT, bo, (float*)d_out, MROWS, D_MODEL, D_MODEL, 1.f);
}
